// MultiHeadSelfAttention_15436112462627
// MI455X (gfx1250) — matrix operand order repaired and checked by reading, not compiled, not run on hardware
//
#include <hip/hip_runtime.h>
#include <hip/hip_bf16.h>

// ---------------------------------------------------------------------------
// Types for CDNA5 WMMA (wave32): v_wmma_f32_16x16x32_bf16
// ---------------------------------------------------------------------------
typedef __attribute__((ext_vector_type(16))) __bf16 v16bf;
typedef __attribute__((ext_vector_type(8)))  float  v8f;

union ABFrag {
    uint4 u[2];   // two contiguous 16B chunks per lane
    v16bf v;
};

__device__ __forceinline__ unsigned short f2bf(float f) {
    unsigned u = __float_as_uint(f);
    unsigned r = u + 0x7FFFu + ((u >> 16) & 1u);   // round-to-nearest-even
    return (unsigned short)(r >> 16);
}
__device__ __forceinline__ float bf2f(unsigned short h) {
    return __uint_as_float(((unsigned)h) << 16);
}

// ---------------------------------------------------------------------------
// Async global->LDS copy (CDNA5 GLOBAL_LOAD_ASYNC_TO_LDS_B128, ASYNCcnt).
// Builtin signature (from hipcc diagnostic): (v4i AS1*, v4i AS3*, imm, imm).
// Guarded so the file compiles even if a toolchain lacks the builtin.
// ---------------------------------------------------------------------------
typedef int v4i_t __attribute__((vector_size(4 * sizeof(int))));

#if defined(__HIP_DEVICE_COMPILE__) && __has_builtin(__builtin_amdgcn_global_load_async_to_lds_b128)
#define HAVE_ASYNC_LDS 1
__device__ __forceinline__ void async_copy16(void* lds_dst, const void* gsrc) {
    __builtin_amdgcn_global_load_async_to_lds_b128(
        (__attribute__((address_space(1))) v4i_t*)gsrc,
        (__attribute__((address_space(3))) v4i_t*)lds_dst,
        0, 0);
}
__device__ __forceinline__ void async_wait() {
#if __has_builtin(__builtin_amdgcn_s_wait_asynccnt)
    __builtin_amdgcn_s_wait_asynccnt(0);
#else
    asm volatile("s_wait_asynccnt 0" ::: "memory");
#endif
}
#else
#define HAVE_ASYNC_LDS 0
__device__ __forceinline__ void async_copy16(void* lds_dst, const void* gsrc) {
    *(uint4*)lds_dst = *(const uint4*)gsrc;   // sync fallback
}
__device__ __forceinline__ void async_wait() {}
#endif

// Problem constants
#define Bsz 4
#define Tsz 2048
#define Dsz 1024
#define Hsz 16
#define HDsz 64

// ---------------------------------------------------------------------------
// 1) f32 -> bf16 elementwise convert
// ---------------------------------------------------------------------------
__global__ __launch_bounds__(256) void cvt_f32_bf16_kernel(
    const float* __restrict__ src, unsigned short* __restrict__ dst, int n) {
    int i = blockIdx.x * 256 + threadIdx.x;
    if (i < n) dst[i] = f2bf(src[i]);
}

// ---------------------------------------------------------------------------
// 2) f32 (K,N) -> bf16 (N,K) transpose-convert (weights)
// ---------------------------------------------------------------------------
__global__ __launch_bounds__(256) void transpose_cvt_kernel(
    const float* __restrict__ W, unsigned short* __restrict__ WT, int K, int N) {
    int i = blockIdx.x * 256 + threadIdx.x;
    if (i < K * N) {
        int k = i / N;
        int n = i - k * N;
        WT[(size_t)n * K + k] = f2bf(W[i]);
    }
}

// ---------------------------------------------------------------------------
// 3) Tiled WMMA GEMM: C(M,N) = A(M,K) * BT(N,K)^T  [bf16 in, f32 accum]
//    Block 256 threads (8 waves, 2x4), tile 128x128, K-step 32,
//    double-buffered LDS with async global->LDS staging.
//    Fragment layouts follow CDNA5 ISA 7.12.2:
//      A-frag lane L: row = L&15, k in {8h..8h+7} U {16+8h..16+8h+7}, h=L>>4
//      B-frag lane L: col = L&15, k in {8h..8h+7} U {16+8h..16+8h+7}
//    -> both are two contiguous 16B LDS reads when A row-major / B transposed.
// ---------------------------------------------------------------------------
template <bool BF16OUT>
__global__ __launch_bounds__(256) void gemm_wmma_kernel(
    const unsigned short* __restrict__ A,   // (M,K) bf16
    const unsigned short* __restrict__ BT,  // (N,K) bf16
    void* __restrict__ Cout,                // (M,N) bf16 or f32
    const float* __restrict__ bias,         // nullptr or (N)
    int M, int N, int K) {
    __shared__ unsigned short As[2][128 * 32];   // 2 x 8 KB
    __shared__ unsigned short Bs[2][128 * 32];   // 2 x 8 KB

    const int tid  = threadIdx.x;
    const int lane = tid & 31;
    const int wave = tid >> 5;
    const int wm = wave >> 2;          // 0..1 -> 64 rows each
    const int wn = wave & 3;           // 0..3 -> 32 cols each
    const int m0 = blockIdx.y * 128;
    const int n0 = blockIdx.x * 128;
    const int mloc = lane & 15;
    const int half = lane >> 4;

    auto load_tile = [&](int buf, int k0) {
#pragma unroll
        for (int c = tid; c < 512; c += 256) {
            int row = c >> 2;          // 4 chunks (64B) per 32-element row
            int cc  = c & 3;
            async_copy16(&((uint4*)As[buf])[c],
                         ((const uint4*)(A  + (size_t)(m0 + row) * K + k0)) + cc);
            async_copy16(&((uint4*)Bs[buf])[c],
                         ((const uint4*)(BT + (size_t)(n0 + row) * K + k0)) + cc);
        }
    };

    v8f acc[4][2];
    const v8f vzero = {0.f, 0.f, 0.f, 0.f, 0.f, 0.f, 0.f, 0.f};
#pragma unroll
    for (int i = 0; i < 4; ++i)
#pragma unroll
        for (int j = 0; j < 2; ++j) acc[i][j] = vzero;

    load_tile(0, 0);
    async_wait();
    __syncthreads();

    for (int k0 = 0; k0 < K; k0 += 32) {
        const int cur = (k0 >> 5) & 1;
        if (k0 + 32 < K) load_tile(cur ^ 1, k0 + 32);   // prefetch next tile

        ABFrag a[4], b[2];
#pragma unroll
        for (int mt = 0; mt < 4; ++mt) {
            int r = wm * 64 + mt * 16 + mloc;
            int off = r * 64 + half * 16;       // bytes into As row
            a[mt].u[0] = *(const uint4*)((const char*)As[cur] + off);
            a[mt].u[1] = *(const uint4*)((const char*)As[cur] + off + 32);
        }
#pragma unroll
        for (int nt = 0; nt < 2; ++nt) {
            int c = wn * 32 + nt * 16 + mloc;
            int off = c * 64 + half * 16;       // bytes into Bs row
            b[nt].u[0] = *(const uint4*)((const char*)Bs[cur] + off);
            b[nt].u[1] = *(const uint4*)((const char*)Bs[cur] + off + 32);
        }
#pragma unroll
        for (int mt = 0; mt < 4; ++mt)
#pragma unroll
            for (int nt = 0; nt < 2; ++nt)
                acc[mt][nt] = __builtin_amdgcn_wmma_f32_16x16x32_bf16(
                    false, a[mt].v, false, b[nt].v, (short)0, acc[mt][nt], false, false);

        async_wait();        // this wave's async writes to the other buffer done
        __syncthreads();     // everyone done reading cur / writing nxt
    }

    // Epilogue: C/D layout -> row = r + 8*half, col = lane&15
#pragma unroll
    for (int mt = 0; mt < 4; ++mt)
#pragma unroll
        for (int nt = 0; nt < 2; ++nt)
#pragma unroll
            for (int r = 0; r < 8; ++r) {
                int row = m0 + wm * 64 + mt * 16 + r + 8 * half;
                int col = n0 + wn * 32 + nt * 16 + mloc;
                float v = acc[mt][nt][r];
                if (BF16OUT) {
                    ((unsigned short*)Cout)[(size_t)row * N + col] = f2bf(v);
                } else {
                    ((float*)Cout)[(size_t)row * N + col] = v + bias[col];
                }
            }
}

// ---------------------------------------------------------------------------
// 4) RoPE + L2-normalize q and k, in place on the bf16 qkv buffer.
//    One wave per (b, t, h, {q|k}) row of 64 elements; lane l owns pair (l, l+32).
// ---------------------------------------------------------------------------
__global__ __launch_bounds__(128) void rope_norm_kernel(unsigned short* __restrict__ qkv) {
    int lane = threadIdx.x & 31;
    int wave = threadIdx.x >> 5;
    int idx  = blockIdx.x * 4 + wave;        // over B*T*H*2
    int qk   = idx & 1;
    int h    = (idx >> 1) & (Hsz - 1);
    int bt   = idx >> 5;                     // b*T + t  (H*2 == 32)
    int t    = bt & (Tsz - 1);

    unsigned short* base = qkv + (size_t)bt * (3 * Dsz) + qk * Dsz + h * HDsz;

    float x1 = bf2f(base[lane]);
    float x2 = bf2f(base[lane + 32]);

    // inv_freq = 10000^(-lane/32)
    float inv_freq = __expf(-((float)lane / 32.0f) * __logf(10000.0f));
    float ang = (float)t * inv_freq;
    float sv = __sinf(ang), cv = __cosf(ang);
    float r1 = x1 * cv - x2 * sv;
    float r2 = x1 * sv + x2 * cv;

    float ss = r1 * r1 + r2 * r2;
#pragma unroll
    for (int d = 16; d >= 1; d >>= 1) ss += __shfl_xor(ss, d);
    float inv = rsqrtf(ss + 1e-6f);

    base[lane]      = f2bf(r1 * inv);
    base[lane + 32] = f2bf(r2 * inv);
}

// ---------------------------------------------------------------------------
// 5) Fused flash attention. Block = 128 threads (4 waves); block covers
//    (b, h, 64 queries), each wave owns a 16-query tile. 32-key blocks are
//    staged in LDS (K via async b128, V transposed), QK^T and PV both via
//    v_wmma_f32_16x16x32_bf16, online softmax in fp32.
// ---------------------------------------------------------------------------
__global__ __launch_bounds__(128) void flash_attn_kernel(
    const unsigned short* __restrict__ qkv,   // (B*T, 3072) bf16, q|k|v
    unsigned short* __restrict__ attn_out) {  // (B*T, 1024) bf16
    __shared__ unsigned short kbuf[32 * 64];      // keys x d      (4 KB)
    __shared__ unsigned short vtbuf[64 * 32];     // d x keys      (4 KB)
    __shared__ unsigned short pbuf[4][16 * 32];   // per-wave P    (4 x 1 KB)

    const int tid  = threadIdx.x;
    const int lane = tid & 31;
    const int wave = tid >> 5;
    const int b  = blockIdx.z;
    const int h  = blockIdx.y;
    const int qb = blockIdx.x * 64 + wave * 16;
    const int m    = lane & 15;
    const int half = lane >> 4;

    // Load this wave's Q fragments once (two 32-wide d-chunks).
    const unsigned short* qrow = qkv + (size_t)(b * Tsz + qb + m) * (3 * Dsz) + h * HDsz;
    ABFrag qf[2];
#pragma unroll
    for (int c = 0; c < 2; ++c) {
        int off = c * 64 + half * 16;             // bytes
        qf[c].u[0] = *(const uint4*)((const char*)qrow + off);
        qf[c].u[1] = *(const uint4*)((const char*)qrow + off + 32);
    }

    float rowm[8], rowl[8];
#pragma unroll
    for (int r = 0; r < 8; ++r) { rowm[r] = -1e30f; rowl[r] = 0.f; }
    v8f o[4];
    const v8f vzero = {0.f, 0.f, 0.f, 0.f, 0.f, 0.f, 0.f, 0.f};
#pragma unroll
    for (int d = 0; d < 4; ++d) o[d] = vzero;

    for (int kb = 0; kb < Tsz; kb += 32) {
        // Stage K block (32 keys x 64 d), natural layout, async b128.
#pragma unroll
        for (int c = tid; c < 256; c += 128) {
            int row = c >> 3;                     // 8 x 16B chunks per row
            int cc  = c & 7;
            async_copy16(&((uint4*)kbuf)[c],
                         ((const uint4*)(qkv + (size_t)(b * Tsz + kb + row) * (3 * Dsz)
                                         + Dsz + h * HDsz)) + cc);
        }
        // Stage V block transposed (d x keys): 2 x b128 global loads/thread.
#pragma unroll
        for (int c = tid; c < 256; c += 128) {
            int row = c >> 3;                     // key index
            int d8  = (c & 7) * 8;                // first d of this chunk
            uint4 t4 = ((const uint4*)(qkv + (size_t)(b * Tsz + kb + row) * (3 * Dsz)
                                       + 2 * Dsz + h * HDsz))[c & 7];
            const unsigned short* e = (const unsigned short*)&t4;
#pragma unroll
            for (int j = 0; j < 8; ++j) vtbuf[(d8 + j) * 32 + row] = e[j];
        }
        async_wait();
        __syncthreads();

        // S tiles: keys [kb,kb+16) and [kb+16,kb+32), contracted over d=64.
        // Load all 4 B-fragments first, then issue the 4 WMMAs.
        ABFrag bk[2][2];
#pragma unroll
        for (int kt = 0; kt < 2; ++kt) {
            int n_loc = kt * 16 + m;
#pragma unroll
            for (int c = 0; c < 2; ++c) {
                int off = n_loc * 128 + c * 64 + half * 16;   // bytes
                bk[kt][c].u[0] = *(const uint4*)((const char*)kbuf + off);
                bk[kt][c].u[1] = *(const uint4*)((const char*)kbuf + off + 32);
            }
        }
        v8f s[2];
        s[0] = vzero; s[1] = vzero;
#pragma unroll
        for (int kt = 0; kt < 2; ++kt)
#pragma unroll
            for (int c = 0; c < 2; ++c)
                s[kt] = __builtin_amdgcn_wmma_f32_16x16x32_bf16(
                    false, qf[c].v, false, bk[kt][c].v, (short)0, s[kt], false, false);

        // Online softmax; per lane we hold stats for rows r + 8*half.
        float al[8];
#pragma unroll
        for (int r = 0; r < 8; ++r) {
            float s0 = s[0][r] * 0.125f;          // 1/sqrt(64)
            float s1 = s[1][r] * 0.125f;
            float mx = fmaxf(s0, s1);
#pragma unroll
            for (int d = 8; d >= 1; d >>= 1) mx = fmaxf(mx, __shfl_xor(mx, d));
            float mnew = fmaxf(rowm[r], mx);
            float p0 = __expf(s0 - mnew);
            float p1 = __expf(s1 - mnew);
            float rs = p0 + p1;
#pragma unroll
            for (int d = 8; d >= 1; d >>= 1) rs += __shfl_xor(rs, d);
            float a = __expf(rowm[r] - mnew);
            rowl[r] = rowl[r] * a + rs;
            rowm[r] = mnew;
            al[r] = a;
            int prow = r + 8 * half;
            pbuf[wave][prow * 32 + m]      = f2bf(p0);
            pbuf[wave][prow * 32 + m + 16] = f2bf(p1);
        }
#pragma unroll
        for (int dt = 0; dt < 4; ++dt)
#pragma unroll
            for (int r = 0; r < 8; ++r) o[dt][r] *= al[r];

        // P fragment (same-wave LDS RAW: DS ops are in-order per wave).
        ABFrag pf;
        {
            int off = m * 64 + half * 16;
            pf.u[0] = *(const uint4*)((const char*)pbuf[wave] + off);
            pf.u[1] = *(const uint4*)((const char*)pbuf[wave] + off + 32);
        }
        // PV: load all 4 V-fragments, then contract 32 keys into 4 d-tiles.
        ABFrag vf[4];
#pragma unroll
        for (int dt = 0; dt < 4; ++dt) {
            int d_loc = dt * 16 + m;
            int off = d_loc * 64 + half * 16;
            vf[dt].u[0] = *(const uint4*)((const char*)vtbuf + off);
            vf[dt].u[1] = *(const uint4*)((const char*)vtbuf + off + 32);
        }
#pragma unroll
        for (int dt = 0; dt < 4; ++dt)
            o[dt] = __builtin_amdgcn_wmma_f32_16x16x32_bf16(
                false, pf.v, false, vf[dt].v, (short)0, o[dt], false, false);
        __syncthreads();
    }

    // Normalize and write (B*T, 1024) bf16.
#pragma unroll
    for (int dt = 0; dt < 4; ++dt)
#pragma unroll
        for (int r = 0; r < 8; ++r) {
            float v = o[dt][r] / rowl[r];
            int t = qb + r + 8 * half;
            attn_out[(size_t)(b * Tsz + t) * Dsz + h * HDsz + dt * 16 + m] = f2bf(v);
        }
}

// ---------------------------------------------------------------------------
// Host-side orchestration
// ---------------------------------------------------------------------------
extern "C" void kernel_launch(void* const* d_in, const int* in_sizes, int n_in,
                              void* d_out, int out_size, void* d_ws, size_t ws_size,
                              hipStream_t stream) {
    (void)in_sizes; (void)n_in; (void)out_size; (void)ws_size;
    const float* x    = (const float*)d_in[0];   // (B,T,D)
    const float* Wqkv = (const float*)d_in[1];   // (D, 3D)
    const float* Wo   = (const float*)d_in[2];   // (D, D)
    const float* bo   = (const float*)d_in[3];   // (D)
    float* out = (float*)d_out;                  // (B,T,D) f32

    char* ws = (char*)d_ws;
    unsigned short* xb    = (unsigned short*)(ws);                        // 16 MB (reused as attn out)
    unsigned short* wqkvT = (unsigned short*)(ws + (16ull << 20));        //  6 MB
    unsigned short* woT   = (unsigned short*)(ws + (22ull << 20));        //  2 MB
    unsigned short* qkvb  = (unsigned short*)(ws + (24ull << 20));        // 48 MB
    unsigned short* attnb = xb;   // x->bf16 no longer needed after GEMM1

    const int M = Bsz * Tsz;      // 8192
    const int K = Dsz;            // 1024
    const int N1 = 3 * Dsz;       // 3072
    const int N2 = Dsz;           // 1024

    // 1) x -> bf16
    cvt_f32_bf16_kernel<<<(M * K) / 256, 256, 0, stream>>>(x, xb, M * K);
    // 2) weights -> bf16, transposed to (N,K)
    transpose_cvt_kernel<<<(K * N1 + 255) / 256, 256, 0, stream>>>(Wqkv, wqkvT, K, N1);
    transpose_cvt_kernel<<<(K * N2 + 255) / 256, 256, 0, stream>>>(Wo, woT, K, N2);
    // 3) qkv = x @ Wqkv   (bf16 out)
    gemm_wmma_kernel<true><<<dim3(N1 / 128, M / 128), 256, 0, stream>>>(
        xb, wqkvT, qkvb, nullptr, M, N1, K);
    // 4) RoPE + L2-normalize q, k in place
    rope_norm_kernel<<<(Bsz * Tsz * Hsz * 2) / 4, 128, 0, stream>>>(qkvb);
    // 5) fused flash attention -> (B*T, D) bf16
    flash_attn_kernel<<<dim3(Tsz / 64, Hsz, Bsz), 128, 0, stream>>>(qkvb, attnb);
    // 6) out = attn @ Wo + bo  (f32 out)
    gemm_wmma_kernel<false><<<dim3(N2 / 128, M / 128), 256, 0, stream>>>(
        attnb, woT, out, bo, M, N2, K);
}
